// ScannedRNN_60438779789392
// MI455X (gfx1250) — matrix operand order repaired and checked by reading, not compiled, not run on hardware
//
#include <hip/hip_runtime.h>
#include <stdint.h>

typedef __attribute__((ext_vector_type(16))) __bf16 v16bf;
typedef __attribute__((ext_vector_type(8)))  float  v8f;

namespace {
constexpr int T_STEPS = 1024;
constexpr int B_TOT   = 256;
constexpr int D_IN    = 128;
constexpr int H_DIM   = 128;
constexpr int G4      = 512;   // 4*H
constexpr int BT      = 16;    // batch rows per workgroup

// LDS layout (dynamic):
//   [0, 256KB)        W packed as B-fragments: ((ntile*8 + kslice)*32 + lane)*16 bf16
//   [+8KB][+8KB]      x stage double buffer (16x128 f32, raw async copy target)
//   [+4KB]            h state, bf16 row-major [16][128]
constexpr unsigned W_BYTES   = 32u * 8u * 32u * 16u * 2u;          // 262144
constexpr unsigned XB_BYTES  = (unsigned)(BT * D_IN * 4);          // 8192
constexpr unsigned H_BYTES   = (unsigned)(BT * H_DIM * 2);         // 4096
constexpr unsigned LDS_TOTAL = W_BYTES + 2u * XB_BYTES + H_BYTES;  // 282624
}

__device__ __forceinline__ float sigmoid_f(float x) {
  return 1.0f / (1.0f + __expf(-x));
}
__device__ __forceinline__ float tanh_f(float x) {
  float e = __expf(2.0f * x);
  return (e - 1.0f) / (e + 1.0f);
}

// One async 16B/lane copy global -> LDS (ASYNCcnt-tracked, CDNA5 async path)
__device__ __forceinline__ void async_copy_b128(unsigned lds_off, uint64_t gaddr) {
  asm volatile("global_load_async_to_lds_b128 %0, %1, off"
               :: "v"(lds_off), "v"(gaddr) : "memory");
}

__global__ __launch_bounds__(256) void lstm_scan_wmma(
    const float* __restrict__ obs,            // [T,B,D]
    const unsigned char* __restrict__ done,   // [T,B] bool bytes
    const float* __restrict__ c0,             // [B,H]
    const float* __restrict__ h0,             // [B,H]
    const float* __restrict__ Wi,             // [D,4H]
    const float* __restrict__ Wh,             // [H,4H]
    const float* __restrict__ bias_p,         // [4H]
    float* __restrict__ out)                  // y [T,B,H] ++ c_f [B,H] ++ h_f [B,H]
{
  extern __shared__ unsigned char lds_raw[];
  __bf16* Wf    = (__bf16*)lds_raw;
  float*  xbuf0 = (float*)(lds_raw + W_BYTES);
  float*  xbuf1 = (float*)(lds_raw + W_BYTES + XB_BYTES);
  __bf16* hbuf  = (__bf16*)(lds_raw + W_BYTES + 2u * XB_BYTES);

  const int tid  = threadIdx.x;
  const int lane = tid & 31;
  const int wv   = tid >> 5;        // wave 0..7: owns h-columns [16*wv, 16*wv+16)
  const int b0   = blockIdx.x * BT; // batch tile base row
  const int lrow = lane & 15;
  const int lhi  = lane >> 4;

  // ---- one-time: pack W = [Wi; Wh] (256 x 512) into bf16 B-fragment order ----
  // B 32x16 bf16 layout: lanes 0-15 -> col n, K = 32s+0..7 | 32s+16..23 ; lanes 16-31: +8
  for (int u = tid; u < 32 * 8 * 32; u += 256) {
    int l = u & 31;
    int s = (u >> 5) & 7;
    int j = u >> 8;                        // ntile 0..31
    int col   = j * 16 + (l & 15);
    int kbase = s * 32 + (l >> 4) * 8;
    __bf16* dst = Wf + (unsigned)u * 16u;
#pragma unroll
    for (int i = 0; i < 16; ++i) {
      int k = kbase + (i < 8 ? i : i + 8);
      float w = (k < D_IN) ? Wi[k * G4 + col] : Wh[(k - D_IN) * G4 + col];
      dst[i] = (__bf16)w;
    }
  }

  // ---- one-time: h0 -> hbuf (bf16), c0 -> carry registers (C layout) ----
  for (int u = tid; u < BT * H_DIM; u += 256)
    hbuf[u] = (__bf16)h0[(b0 + (u >> 7)) * H_DIM + (u & 127)];

  v8f c;
  float hreg[8];
#pragma unroll
  for (int r = 0; r < 8; ++r) {
    int row = r + 8 * lhi, col = wv * 16 + lrow;
    c[r]    = c0[(b0 + row) * H_DIM + col];
    hreg[r] = h0[(b0 + row) * H_DIM + col];
  }

  float bgate[4];
#pragma unroll
  for (int g = 0; g < 4; ++g) bgate[g] = bias_p[g * H_DIM + wv * 16 + lrow];

  // ---- prologue: async-stage x(t=0) into xbuf0 (2 x b128 per thread) ----
  {
    uint64_t ga = (uint64_t)(uintptr_t)(obs + (size_t)b0 * D_IN) + (unsigned)tid * 16u;
    unsigned lo = (unsigned)(uintptr_t)xbuf0 + (unsigned)tid * 16u;
    async_copy_b128(lo, ga);
    async_copy_b128(lo + 4096u, ga + 4096u);
  }

#pragma unroll 1
  for (int t = 0; t < T_STEPS; ++t) {
    const float* curx = (t & 1) ? xbuf1 : xbuf0;
    float*       nxtx = (t & 1) ? xbuf0 : xbuf1;

    // wait this wave's async copies for curx, then make all waves' copies +
    // previous step's hbuf writes visible
    asm volatile("s_wait_asynccnt 0x0" ::: "memory");
    __syncthreads();

    // kick off x(t+1) copy; overlaps the whole compute of step t.
    // (nxtx was last read in step t-1, all reads fenced by that step's barrier)
    if (t + 1 < T_STEPS) {
      uint64_t ga = (uint64_t)(uintptr_t)(obs + ((size_t)(t + 1) * B_TOT + b0) * D_IN)
                  + (unsigned)tid * 16u;
      unsigned lo = (unsigned)(uintptr_t)nxtx + (unsigned)tid * 16u;
      async_copy_b128(lo, ga);
      async_copy_b128(lo + 4096u, ga + 4096u);
    }

    // ---- done masking (reset carry BEFORE the gemm, per reference) ----
    const size_t dbase = (size_t)t * B_TOT + b0;
    const unsigned char dn_row = done[dbase + lrow];   // for this lane's A rows
    {
      uint64_t dbits = *(const uint64_t*)(done + dbase + (size_t)lhi * 8);
#pragma unroll
      for (int r = 0; r < 8; ++r)
        if ((dbits >> (8 * r)) & 0xffu) c[r] = 0.0f;
    }

    // ---- A fragments: x (f32->bf16 on the fly) and h (bf16, done-masked) ----
    // A 16x32 bf16 layout: lanes 0-15: K = 32s+0..7 | 32s+16..23 ; lanes 16-31: +8
    union AV { v16bf v; __bf16 e[16]; };
    AV ax[4], ah[4];
    {
      const float*  xr = curx + lrow * D_IN;
      const __bf16* hr = hbuf + lrow * H_DIM;
#pragma unroll
      for (int s = 0; s < 4; ++s) {
        int e0 = 32 * s + lhi * 8;
        int e1 = e0 + 16;
#pragma unroll
        for (int i = 0; i < 8; ++i) {
          ax[s].e[i]     = (__bf16)xr[e0 + i];
          ax[s].e[8 + i] = (__bf16)xr[e1 + i];
          ah[s].e[i]     = dn_row ? (__bf16)0.0f : hr[e0 + i];
          ah[s].e[8 + i] = dn_row ? (__bf16)0.0f : hr[e1 + i];
        }
      }
    }

    // ---- gates = [x|h] @ [Wi;Wh] + b : 4 tiles x 8 WMMAs (K=256) ----
    v8f acc[4];
#pragma unroll
    for (int g = 0; g < 4; ++g) {
      v8f a;
#pragma unroll
      for (int r = 0; r < 8; ++r) a[r] = bgate[g];
      const int jt = g * 8 + wv;  // column tile in [0,32)
#pragma unroll
      for (int s = 0; s < 4; ++s) {
        v16bf bm = *(const v16bf*)(Wf + (((unsigned)jt * 8u + s) * 32u + lane) * 16u);
        a = __builtin_amdgcn_wmma_f32_16x16x32_bf16(false, ax[s].v, false, bm,
                                                    (short)0, a, false, false);
      }
#pragma unroll
      for (int s = 4; s < 8; ++s) {
        v16bf bm = *(const v16bf*)(Wf + (((unsigned)jt * 8u + s) * 32u + lane) * 16u);
        a = __builtin_amdgcn_wmma_f32_16x16x32_bf16(false, ah[s - 4].v, false, bm,
                                                    (short)0, a, false, false);
      }
      acc[g] = a;
    }

    // all waves finished reading hbuf/curx before hbuf is overwritten
    __syncthreads();

    // ---- nonlinearity + carry update + outputs ----
    const size_t ybase = ((size_t)t * B_TOT + b0) * H_DIM;
    const int col = wv * 16 + lrow;
#pragma unroll
    for (int r = 0; r < 8; ++r) {
      float ig = sigmoid_f(acc[0][r]);
      float fg = sigmoid_f(acc[1][r]);
      float gg = tanh_f(acc[2][r]);
      float og = sigmoid_f(acc[3][r]);
      float cn = fg * c[r] + ig * gg;
      float hn = og * tanh_f(cn);
      c[r]    = cn;
      hreg[r] = hn;
      int row = r + 8 * lhi;
      hbuf[row * H_DIM + col]            = (__bf16)hn;  // next step's A operand
      out[ybase + (size_t)row * H_DIM + col] = hn;      // y[t]
    }
  }

  // ---- final carries: (c_f, h_f) appended after y ----
  float* cf = out + (size_t)T_STEPS * B_TOT * H_DIM;
  float* hf = cf + (size_t)B_TOT * H_DIM;
#pragma unroll
  for (int r = 0; r < 8; ++r) {
    int row = r + 8 * lhi, col = wv * 16 + lrow;
    cf[(b0 + row) * H_DIM + col] = c[r];
    hf[(b0 + row) * H_DIM + col] = hreg[r];
  }
}

extern "C" void kernel_launch(void* const* d_in, const int* in_sizes, int n_in,
                              void* d_out, int out_size, void* d_ws, size_t ws_size,
                              hipStream_t stream) {
  (void)in_sizes; (void)n_in; (void)out_size; (void)d_ws; (void)ws_size;
  const float*         obs  = (const float*)d_in[0];
  const unsigned char* done = (const unsigned char*)d_in[1];
  const float*         c0   = (const float*)d_in[2];
  const float*         h0   = (const float*)d_in[3];
  const float*         Wi   = (const float*)d_in[4];
  const float*         Wh   = (const float*)d_in[5];
  const float*         b    = (const float*)d_in[6];
  float*               out  = (float*)d_out;

  lstm_scan_wmma<<<dim3(B_TOT / BT), dim3(256), LDS_TOTAL, stream>>>(
      obs, done, c0, h0, Wi, Wh, b, out);
}
